// QKVAttention_63101659513393
// MI455X (gfx1250) — matrix operand order repaired and checked by reading, not compiled, not run on hardware
//
#include <hip/hip_runtime.h>

#define B_ 2
#define L_ 2048
#define D_ 1024
#define H_ 16
#define DH_ 64
#define M_TOT (B_ * L_)  // 4096

typedef __attribute__((ext_vector_type(16))) __bf16 v16bf;
typedef __attribute__((ext_vector_type(8)))  float  v8f;

union BF16Frag { v16bf v; unsigned short u[16]; };

__device__ __forceinline__ unsigned short f2bf(float f) {
  unsigned int u = __float_as_uint(f);
  u += 0x7FFFu + ((u >> 16) & 1u);  // round-to-nearest-even
  return (unsigned short)(u >> 16);
}

__device__ __forceinline__ v8f v8f_zero() {
  v8f z;
#pragma unroll
  for (int i = 0; i < 8; ++i) z[i] = 0.0f;
  return z;
}

__device__ __forceinline__ v8f wmma_bf16(v16bf a, v16bf b, v8f c) {
  return __builtin_amdgcn_wmma_f32_16x16x32_bf16(false, a, false, b, (short)0, c,
                                                 false, false);
}

// ---- CDNA5 async global->LDS copy (ASYNCcnt) -------------------------------
// LDS dest address = wave-relative LDS byte offset (low 32 bits of generic
// pointer to a __shared__ object). 16 bytes per lane per instruction.
__device__ __forceinline__ unsigned lds_lo32(const void* p) {
  return (unsigned)(unsigned long long)p;
}
__device__ __forceinline__ void async_b128(unsigned lds, const void* g) {
  unsigned long long ga = (unsigned long long)g;
  asm volatile("global_load_async_to_lds_b128 %0, %1, off"
               :: "v"(lds), "v"(ga) : "memory");
}
__device__ __forceinline__ void wait_async0() {
  asm volatile("s_wait_asynccnt 0" ::: "memory");
}
// copy 64 bytes (32 ushorts) for this thread: 4 async b128 ops
__device__ __forceinline__ void stage64B_async(unsigned ldsbase,
                                               const unsigned short* g) {
#pragma unroll
  for (int i = 0; i < 4; ++i) async_b128(ldsbase + 16u * i, g + 8 * i);
}

// ---- fragment loaders (wave32; k = 16*(lane/16) + e in A and B alike) ----------------------------
__device__ __forceinline__ v16bf load_A_frag(const unsigned short* lds, int row0,
                                             int stride, int k0) {
  const int lane = threadIdx.x & 31;
  BF16Frag f;
  const int m = row0 + (lane & 15);
  const int kh = (lane >> 4) << 4;
#pragma unroll
  for (int e = 0; e < 16; ++e) {
    int k = e + kh;
    f.u[e] = lds[m * stride + k0 + k];
  }
  return f.v;
}

// B[k][n] = lds[n0+n][k0+k]  (A @ B^T form)
__device__ __forceinline__ v16bf load_B_frag_nk(const unsigned short* lds, int n0,
                                                int stride, int k0) {
  const int lane = threadIdx.x & 31;
  BF16Frag f;
  const int n = n0 + (lane & 15);
  const int kh = (lane >> 4) << 4;
#pragma unroll
  for (int e = 0; e < 16; ++e) f.u[e] = lds[n * stride + k0 + kh + e];
  return f.v;
}

// B[k][n] = lds[k0+k][n0+n]  (natural K-major)
__device__ __forceinline__ v16bf load_B_frag_kn(const unsigned short* lds, int k0,
                                                int stride, int n0) {
  const int lane = threadIdx.x & 31;
  BF16Frag f;
  const int n = n0 + (lane & 15);
  const int kh = (lane >> 4) << 4;
#pragma unroll
  for (int e = 0; e < 16; ++e) f.u[e] = lds[(k0 + kh + e) * stride + n];
  return f.v;
}

// ---------------------------------------------------------------- conversion
__global__ void cvt_f32_bf16(const float* __restrict__ in,
                             unsigned short* __restrict__ out, int n4) {
  int i = blockIdx.x * blockDim.x + threadIdx.x;
  if (i < n4) {
    float4 f = reinterpret_cast<const float4*>(in)[i];
    uint2 r;
    r.x = (unsigned)f2bf(f.x) | ((unsigned)f2bf(f.y) << 16);
    r.y = (unsigned)f2bf(f.z) | ((unsigned)f2bf(f.w) << 16);
    reinterpret_cast<uint2*>(out)[i] = r;
  }
}

// ------------------------------------------------------------------- GEMM
// C[M,N] = (A[M,K] @ B[N,K]^T + bias[n]) * scale ; bf16 in, f32 accum.
// 64x64 block tile, 4 waves, 64-wide K chunks, double-buffered LDS filled by
// CDNA5 async global->LDS copies overlapped with WMMA compute.
template <int OUTF>
__global__ __launch_bounds__(128) void gemm_bt_kernel(
    const unsigned short* __restrict__ A, const unsigned short* __restrict__ Bw,
    const float* __restrict__ bias, unsigned short* __restrict__ Cbf,
    float* __restrict__ Cf, int M, int N, int K, float scale) {
  __shared__ unsigned short lA[2][64][72];
  __shared__ unsigned short lB[2][64][72];

  const int tid = threadIdx.x;
  const int wave = tid >> 5;
  const int lane = tid & 31;
  const int m0 = blockIdx.y * 64;
  const int n0 = blockIdx.x * 64;

  v8f acc[4];
#pragma unroll
  for (int t = 0; t < 4; ++t) acc[t] = v8f_zero();

  const int r = tid >> 1;         // 0..63
  const int c0 = (tid & 1) << 5;  // 0 or 32 ushorts
  const unsigned short* gArow = A + (size_t)(m0 + r) * K + c0;
  const unsigned short* gBrow = Bw + (size_t)(n0 + r) * K + c0;
  const unsigned ldsA0 = lds_lo32(&lA[0][r][c0]);
  const unsigned ldsB0 = lds_lo32(&lB[0][r][c0]);
  const unsigned bufstep = (unsigned)sizeof(lA) / 2;  // bytes between buffers

  // preload chunk 0 into buffer 0
  stage64B_async(ldsA0, gArow);
  stage64B_async(ldsB0, gBrow);
  wait_async0();
  __syncthreads();

  const int niter = K >> 6;
  int cur = 0;
  for (int it = 0; it < niter; ++it) {
    if (it + 1 < niter) {  // fire-and-forget next chunk into other buffer
      const unsigned nb = (unsigned)(cur ^ 1) * bufstep;
      stage64B_async(ldsA0 + nb, gArow + (it + 1) * 64);
      stage64B_async(ldsB0 + nb, gBrow + (it + 1) * 64);
    }

    const unsigned short* sA = &lA[cur][0][0];
    const unsigned short* sB = &lB[cur][0][0];
#pragma unroll
    for (int ks = 0; ks < 2; ++ks) {
      v16bf af = load_A_frag(sA, wave * 16, 72, ks * 32);
      v16bf bf0 = load_B_frag_nk(sB, 0, 72, ks * 32);
      v16bf bf1 = load_B_frag_nk(sB, 16, 72, ks * 32);
      v16bf bf2 = load_B_frag_nk(sB, 32, 72, ks * 32);
      v16bf bf3 = load_B_frag_nk(sB, 48, 72, ks * 32);
      acc[0] = wmma_bf16(af, bf0, acc[0]);
      acc[1] = wmma_bf16(af, bf1, acc[1]);
      acc[2] = wmma_bf16(af, bf2, acc[2]);
      acc[3] = wmma_bf16(af, bf3, acc[3]);
    }

    wait_async0();
    __syncthreads();
    cur ^= 1;
  }

  const int mh = (lane >> 4) << 3;
#pragma unroll
  for (int nt = 0; nt < 4; ++nt) {
    const int n = n0 + nt * 16 + (lane & 15);
    const float bsv = bias[n];
#pragma unroll
    for (int j = 0; j < 8; ++j) {
      const int m = m0 + wave * 16 + j + mh;
      const float val = (acc[nt][j] + bsv) * scale;
      const size_t idx = (size_t)m * N + n;
      if constexpr (OUTF) Cf[idx] = val;
      else Cbf[idx] = f2bf(val);
    }
  }
}

// --------------------------------------------------------- flash attention
// One WG (4 waves) = 64 query rows of one (b,h). 64-key blocks, K/V double-
// buffered via async global->LDS copies overlapped with S/softmax/PV compute.
__global__ __launch_bounds__(128) void attn_kernel(
    const unsigned short* __restrict__ qp, const unsigned short* __restrict__ kp,
    const unsigned short* __restrict__ vp, unsigned short* __restrict__ ao) {
  __shared__ unsigned short lQ[64][72];
  __shared__ unsigned short lK[2][64][72];
  __shared__ unsigned short lV[2][64][72];
  __shared__ unsigned short lP[4][16][72];

  const int tid = threadIdx.x;
  const int wave = tid >> 5;
  const int lane = tid & 31;
  const int q0 = blockIdx.x * 64;
  const int b = blockIdx.y >> 4;
  const int h = blockIdx.y & 15;
  const size_t colbase = (size_t)h * DH_;

  const int r = tid >> 1;         // 0..63
  const int c0 = (tid & 1) << 5;  // 0 or 32

  const size_t growbase = (size_t)(b * L_ + r) * D_ + colbase + c0;
  const unsigned ldsK0 = lds_lo32(&lK[0][r][c0]);
  const unsigned ldsV0 = lds_lo32(&lV[0][r][c0]);
  const unsigned bufstep = (unsigned)sizeof(lK) / 2;

  // stage Q tile + K/V block 0 asynchronously
  stage64B_async(lds_lo32(&lQ[r][c0]),
                 qp + (size_t)(b * L_ + q0 + r) * D_ + colbase + c0);
  stage64B_async(ldsK0, kp + growbase);
  stage64B_async(ldsV0, vp + growbase);
  wait_async0();
  __syncthreads();

  v16bf aq0 = load_A_frag(&lQ[0][0], wave * 16, 72, 0);
  v16bf aq1 = load_A_frag(&lQ[0][0], wave * 16, 72, 32);

  v8f o[4];
#pragma unroll
  for (int t = 0; t < 4; ++t) o[t] = v8f_zero();
  float mrow[8], lrow[8];
#pragma unroll
  for (int j = 0; j < 8; ++j) {
    mrow[j] = -__builtin_inff();
    lrow[j] = 0.0f;
  }

  const int mh = (lane >> 4) << 3;
  const int nblk = L_ / 64;
  int cur = 0;
  for (int kb = 0; kb < nblk; ++kb) {
    if (kb + 1 < nblk) {  // async-stage next K/V block into other buffer
      const unsigned nb = (unsigned)(cur ^ 1) * bufstep;
      const size_t gnext = growbase + (size_t)(kb + 1) * 64 * D_;
      stage64B_async(ldsK0 + nb, kp + gnext);
      stage64B_async(ldsV0 + nb, vp + gnext);
    }

    // S[16 x 64] = Q[16 x 64] @ Kblk^T[64 x 64]
    const unsigned short* sK = &lK[cur][0][0];
    v8f s[4];
#pragma unroll
    for (int nt = 0; nt < 4; ++nt) s[nt] = v8f_zero();
#pragma unroll
    for (int nt = 0; nt < 4; ++nt) {
      v16bf bk0 = load_B_frag_nk(sK, nt * 16, 72, 0);
      v16bf bk1 = load_B_frag_nk(sK, nt * 16, 72, 32);
      s[nt] = wmma_bf16(aq0, bk0, s[nt]);
      s[nt] = wmma_bf16(aq1, bk1, s[nt]);
    }

    // online softmax: row M=j+8*(lane>=16), cols across 16-lane halves
    float alpha[8];
#pragma unroll
    for (int j = 0; j < 8; ++j) {
      float bmax = fmaxf(fmaxf(s[0][j], s[1][j]), fmaxf(s[2][j], s[3][j]));
#pragma unroll
      for (int off = 1; off < 16; off <<= 1)
        bmax = fmaxf(bmax, __shfl_xor(bmax, off));
      const float mn = fmaxf(mrow[j], bmax);
      alpha[j] = __expf(mrow[j] - mn);
      mrow[j] = mn;
      float rs = 0.0f;
#pragma unroll
      for (int nt = 0; nt < 4; ++nt) {
        const float p = __expf(s[nt][j] - mn);
        rs += p;
        lP[wave][j + mh][(lane & 15) + 16 * nt] = f2bf(p);
      }
#pragma unroll
      for (int off = 1; off < 16; off <<= 1) rs += __shfl_xor(rs, off);
      lrow[j] = lrow[j] * alpha[j] + rs;
#pragma unroll
      for (int nt = 0; nt < 4; ++nt) o[nt][j] *= alpha[j];
    }
    __syncthreads();  // P visible (C-layout -> A-layout via LDS)

    const unsigned short* sV = &lV[cur][0][0];
    v16bf ap0 = load_A_frag(&lP[wave][0][0], 0, 72, 0);
    v16bf ap1 = load_A_frag(&lP[wave][0][0], 0, 72, 32);
#pragma unroll
    for (int nt = 0; nt < 4; ++nt) {
      v16bf bv0 = load_B_frag_kn(sV, 0, 72, nt * 16);
      v16bf bv1 = load_B_frag_kn(sV, 32, 72, nt * 16);
      o[nt] = wmma_bf16(ap0, bv0, o[nt]);
      o[nt] = wmma_bf16(ap1, bv1, o[nt]);
    }

    wait_async0();    // own async copies for next block landed
    __syncthreads();  // all waves: done reading cur, next buffer complete
    cur ^= 1;
  }

#pragma unroll
  for (int j = 0; j < 8; ++j) {
    const float inv = 1.0f / lrow[j];
    const int m = q0 + wave * 16 + j + mh;
#pragma unroll
    for (int nt = 0; nt < 4; ++nt) {
      const int c = nt * 16 + (lane & 15);
      ao[(size_t)(b * L_ + m) * D_ + colbase + c] = f2bf(o[nt][j] * inv);
    }
  }
}

// ---------------------------------------------------------------- launcher
extern "C" void kernel_launch(void* const* d_in, const int* in_sizes, int n_in,
                              void* d_out, int out_size, void* d_ws,
                              size_t ws_size, hipStream_t stream) {
  const float* q = (const float*)d_in[0];
  const float* k = (const float*)d_in[1];
  const float* v = (const float*)d_in[2];
  const float* Wq = (const float*)d_in[3];
  const float* bq = (const float*)d_in[4];
  const float* Wk = (const float*)d_in[5];
  const float* bk = (const float*)d_in[6];
  const float* Wv = (const float*)d_in[7];
  const float* bv = (const float*)d_in[8];
  const float* Wo = (const float*)d_in[9];
  const float* bo = (const float*)d_in[10];
  float* out = (float*)d_out;

  char* ws = (char*)d_ws;
  const size_t MB = (size_t)1 << 20;
  unsigned short* qb = (unsigned short*)(ws + 0 * MB);
  unsigned short* kb = (unsigned short*)(ws + 8 * MB);
  unsigned short* vb = (unsigned short*)(ws + 16 * MB);
  unsigned short* wqb = (unsigned short*)(ws + 24 * MB);
  unsigned short* wkb = (unsigned short*)(ws + 26 * MB);
  unsigned short* wvb = (unsigned short*)(ws + 28 * MB);
  unsigned short* wob = (unsigned short*)(ws + 30 * MB);
  unsigned short* qpb = (unsigned short*)(ws + 32 * MB);
  unsigned short* kpb = (unsigned short*)(ws + 40 * MB);
  unsigned short* vpb = (unsigned short*)(ws + 48 * MB);
  unsigned short* aob = (unsigned short*)(ws + 56 * MB);

  const int nAct = M_TOT * D_;  // 4194304
  const int nW = D_ * D_;       // 1048576
  cvt_f32_bf16<<<dim3(nAct / 4 / 256), dim3(256), 0, stream>>>(q, qb, nAct / 4);
  cvt_f32_bf16<<<dim3(nAct / 4 / 256), dim3(256), 0, stream>>>(k, kb, nAct / 4);
  cvt_f32_bf16<<<dim3(nAct / 4 / 256), dim3(256), 0, stream>>>(v, vb, nAct / 4);
  cvt_f32_bf16<<<dim3(nW / 4 / 256), dim3(256), 0, stream>>>(Wq, wqb, nW / 4);
  cvt_f32_bf16<<<dim3(nW / 4 / 256), dim3(256), 0, stream>>>(Wk, wkb, nW / 4);
  cvt_f32_bf16<<<dim3(nW / 4 / 256), dim3(256), 0, stream>>>(Wv, wvb, nW / 4);
  cvt_f32_bf16<<<dim3(nW / 4 / 256), dim3(256), 0, stream>>>(Wo, wob, nW / 4);

  dim3 ggrid(D_ / 64, M_TOT / 64);  // (16, 64)
  dim3 gblk(128);
  const float scale = 0.125f;  // DH^-0.5
  gemm_bt_kernel<0><<<ggrid, gblk, 0, stream>>>(qb, wqb, bq, qpb, nullptr,
                                                M_TOT, D_, D_, scale);
  gemm_bt_kernel<0><<<ggrid, gblk, 0, stream>>>(kb, wkb, bk, kpb, nullptr,
                                                M_TOT, D_, D_, 1.0f);
  gemm_bt_kernel<0><<<ggrid, gblk, 0, stream>>>(vb, wvb, bv, vpb, nullptr,
                                                M_TOT, D_, D_, 1.0f);

  attn_kernel<<<dim3(L_ / 64, B_ * H_), gblk, 0, stream>>>(qpb, kpb, vpb, aob);

  gemm_bt_kernel<1><<<ggrid, gblk, 0, stream>>>(aob, wob, bo, nullptr, out,
                                                M_TOT, D_, D_, 1.0f);
}
